// GCN_51247549776296
// MI455X (gfx1250) — compile-verified, matrix operand order checked by reading, not run on hardware
//
#include <hip/hip_runtime.h>
#include <hip/hip_bf16.h>
#include <math.h>

#define N_NODES  50000
#define N_EDGES  600000
#define N_FEAT   128
#define EMB      128
#define N_GRAPHS 1024
#define ROW_TILES (N_NODES / 16)   // 3125, exact

typedef __attribute__((ext_vector_type(16))) __bf16 v16bf;
typedef __attribute__((ext_vector_type(2)))  __bf16 bf16x2;
typedef __attribute__((ext_vector_type(2)))  float  f32x2;
typedef __attribute__((ext_vector_type(8)))  float  v8f;

union FragA {            // pair i occupies 32-bit VGPR i: elements {2i, 2i+1}
    v16bf  v;
    bf16x2 p[8];
};

__device__ __forceinline__ bf16x2 cvt_pk(float a, float b) {
    f32x2 t = { a, b };
    return __builtin_convertvector(t, bf16x2);   // single v_cvt_pk_bf16_f32
}

__global__ void zero_f32(float* __restrict__ p, int n) {
    int i = blockIdx.x * blockDim.x + threadIdx.x;
    if (i < n) p[i] = 0.0f;
}

// ---------- degree / normalization ----------
__global__ void deg_kernel(const int* __restrict__ dst, float* __restrict__ deg, int nE) {
    int e = blockIdx.x * blockDim.x + threadIdx.x;
    if (e < nE) atomicAdd(&deg[dst[e]], 1.0f);
}

__global__ void dinv_kernel(float* __restrict__ deg, int n) {
    int i = blockIdx.x * blockDim.x + threadIdx.x;
    if (i < n) deg[i] = rsqrtf(deg[i] + 1.0f);   // in place: deg -> dinv
}

// ---------- dense GEMM: C[M,128] = (relu?)A[M,128] @ W[128,128], bf16 WMMA ----------
template<bool RELU>
__global__ __launch_bounds__(256) void gemm_wmma_bf16(
    const float* __restrict__ A, const float* __restrict__ W,
    float* __restrict__ C)
{
    __shared__ __bf16 Wt[128][136];   // Wt[n][k], padded: stride 272B spreads banks

    int tid = threadIdx.x;
    // Load + transpose W (f32, row-major [k][n]) into LDS as bf16 (one-time per block)
    for (int idx = tid; idx < 128 * 128; idx += 256) {
        int k = idx >> 7, n = idx & 127;
        Wt[n][k] = (__bf16)W[idx];
    }
    __syncthreads();

    const int lane = tid & 31;
    const int wave = tid >> 5;
    const int mn   = lane & 15;    // A: row within tile | B/C: column within tile
    const int half = lane >> 4;    // K-half selector per ISA VGPR layout

    for (int tile = blockIdx.x * 8 + wave; tile < ROW_TILES; tile += gridDim.x * 8) {
        const int row0 = tile * 16;
        const float4* __restrict__ a4 =
            (const float4*)(A + (size_t)(row0 + mn) * 128);

        // Build A fragments for all 4 K-steps (16x32 bf16 layout, ISA 7.12.2)
        FragA afrag[4];
        #pragma unroll
        for (int ks = 0; ks < 4; ++ks) {
            const int kb4 = (ks * 32 + half * 8) >> 2;   // float4 index, even
            float4 q0 = a4[kb4];         // K = kb .. kb+3
            float4 q1 = a4[kb4 + 1];     // K = kb+4 .. kb+7
            float4 q2 = a4[kb4 + 4];     // K = kb+16 .. kb+19
            float4 q3 = a4[kb4 + 5];     // K = kb+20 .. kb+23
            if (RELU) {
                q0.x = fmaxf(q0.x, 0.f); q0.y = fmaxf(q0.y, 0.f);
                q0.z = fmaxf(q0.z, 0.f); q0.w = fmaxf(q0.w, 0.f);
                q1.x = fmaxf(q1.x, 0.f); q1.y = fmaxf(q1.y, 0.f);
                q1.z = fmaxf(q1.z, 0.f); q1.w = fmaxf(q1.w, 0.f);
                q2.x = fmaxf(q2.x, 0.f); q2.y = fmaxf(q2.y, 0.f);
                q2.z = fmaxf(q2.z, 0.f); q2.w = fmaxf(q2.w, 0.f);
                q3.x = fmaxf(q3.x, 0.f); q3.y = fmaxf(q3.y, 0.f);
                q3.z = fmaxf(q3.z, 0.f); q3.w = fmaxf(q3.w, 0.f);
            }
            // elements 0..7 = K lo-half, 8..15 = K hi-half; one packed cvt per VGPR
            afrag[ks].p[0] = cvt_pk(q0.x, q0.y);
            afrag[ks].p[1] = cvt_pk(q0.z, q0.w);
            afrag[ks].p[2] = cvt_pk(q1.x, q1.y);
            afrag[ks].p[3] = cvt_pk(q1.z, q1.w);
            afrag[ks].p[4] = cvt_pk(q2.x, q2.y);
            afrag[ks].p[5] = cvt_pk(q2.z, q2.w);
            afrag[ks].p[6] = cvt_pk(q3.x, q3.y);
            afrag[ks].p[7] = cvt_pk(q3.z, q3.w);
        }

        // Sweep the 8 column tiles of the 16x128 output slab
        #pragma unroll
        for (int nt = 0; nt < 8; ++nt) {
            v8f acc = {};
            #pragma unroll
            for (int ks = 0; ks < 4; ++ks) {
                const __bf16* __restrict__ bp = &Wt[nt * 16 + mn][ks * 32 + half * 8];
                v16bf bfrag;
                #pragma unroll
                for (int j = 0; j < 8; ++j) {
                    bfrag[j]     = bp[j];        // 16B contiguous -> ds_load_b128
                    bfrag[8 + j] = bp[16 + j];
                }
                acc = __builtin_amdgcn_wmma_f32_16x16x32_bf16(
                        false, afrag[ks].v, false, bfrag, (short)0, acc, false, false);
            }
            // C/D layout: lane -> (row = r + 8*half, col = lane&15)
            float* __restrict__ crow = C + (size_t)row0 * 128 + nt * 16;
            #pragma unroll
            for (int r = 0; r < 8; ++r)
                crow[(size_t)(r + 8 * half) * 128 + mn] = acc[r];
        }
    }
}

// ---------- agg init: agg = h * dinv^2 + bias (self-loop + bias) ----------
__global__ void init_agg(const float* __restrict__ h, const float* __restrict__ dinv,
                         const float* __restrict__ bias, float* __restrict__ agg)
{
    int tid = blockIdx.x * blockDim.x + threadIdx.x;     // one float4 each
    if (tid >= N_NODES * 32) return;
    int node = tid >> 5, q = tid & 31;
    float di = dinv[node];
    float sl = di * di;
    float4 v = ((const float4*)(h + (size_t)node * 128))[q];
    float4 b = ((const float4*)bias)[q];
    float4 o = { v.x * sl + b.x, v.y * sl + b.y, v.z * sl + b.z, v.w * sl + b.w };
    ((float4*)(agg + (size_t)node * 128))[q] = o;
}

// ---------- edge scatter: agg[dst] += h[src] * dinv[src]*dinv[dst] ----------
__global__ void edge_scatter(const int* __restrict__ src, const int* __restrict__ dst,
                             const float* __restrict__ h, const float* __restrict__ dinv,
                             float* __restrict__ agg)
{
    int tid = blockIdx.x * blockDim.x + threadIdx.x;
    int e = tid >> 5;                          // one wave per edge
    if (e >= N_EDGES) return;
    int q = tid & 31;                          // 4 features per lane
    int s = src[e], d = dst[e];
    float norm = dinv[s] * dinv[d];
    float4 v = ((const float4*)(h + (size_t)s * 128))[q];
    float* ap = agg + (size_t)d * 128 + q * 4;
    atomicAdd(ap + 0, v.x * norm);
    atomicAdd(ap + 1, v.y * norm);
    atomicAdd(ap + 2, v.z * norm);
    atomicAdd(ap + 3, v.w * norm);
}

// ---------- pooling ----------
__global__ void counts_kernel(const int* __restrict__ batch, float* __restrict__ counts) {
    int i = blockIdx.x * blockDim.x + threadIdx.x;
    if (i < N_NODES) atomicAdd(&counts[batch[i]], 1.0f);
}

__global__ void pool_kernel(const float* __restrict__ agg, const int* __restrict__ batch,
                            unsigned int* __restrict__ gmax_u, float* __restrict__ gsum)
{
    int tid = blockIdx.x * blockDim.x + threadIdx.x;
    if (tid >= N_NODES * 128) return;
    int node = tid >> 7, f = tid & 127;
    float v = fmaxf(agg[tid], 0.0f);           // fused ReLU of layer 2
    int b = batch[node];
    atomicMax(&gmax_u[b * 128 + f], __float_as_uint(v));  // valid: v >= 0
    atomicAdd(&gsum[b * 128 + f], v);
}

// ---------- MLP head: [G,256] -> 128 -> 128 -> 1, sigmoid ----------
__global__ __launch_bounds__(128) void mlp_kernel(
    const float* __restrict__ gmax, const float* __restrict__ gsum,
    const float* __restrict__ counts,
    const float* __restrict__ Wl1, const float* __restrict__ bl1,
    const float* __restrict__ Wl2, const float* __restrict__ bl2,
    const float* __restrict__ Wl3, const float* __restrict__ bl3,
    float* __restrict__ out)
{
    __shared__ float g[256];
    __shared__ float o1[128];
    __shared__ float red[4];
    const int gi = blockIdx.x;
    const int t  = threadIdx.x;

    float cnt = fmaxf(counts[gi], 1.0f);
    g[t]       = gmax[gi * 128 + t];
    g[128 + t] = gsum[gi * 128 + t] / cnt;
    __syncthreads();

    float acc = bl1[t];
    #pragma unroll 8
    for (int j = 0; j < 256; ++j) acc += g[j] * Wl1[j * 128 + t];
    o1[t] = fmaxf(acc, 0.0f);
    __syncthreads();

    acc = bl2[t];
    #pragma unroll 8
    for (int j = 0; j < 128; ++j) acc += o1[j] * Wl2[j * 128 + t];
    float o2 = fmaxf(acc, 0.0f);

    float p = o2 * Wl3[t];
    #pragma unroll
    for (int off = 16; off > 0; off >>= 1) p += __shfl_down(p, off, 32);
    if ((t & 31) == 0) red[t >> 5] = p;
    __syncthreads();
    if (t == 0) {
        float s = red[0] + red[1] + red[2] + red[3] + bl3[0];
        out[gi] = 1.0f / (1.0f + expf(-s));
    }
}

// ---------- launcher ----------
extern "C" void kernel_launch(void* const* d_in, const int* in_sizes, int n_in,
                              void* d_out, int out_size, void* d_ws, size_t ws_size,
                              hipStream_t stream)
{
    const float* x    = (const float*)d_in[0];
    const int*   eidx = (const int*)d_in[1];
    const int*   src  = eidx;
    const int*   dst  = eidx + N_EDGES;
    const int*   batch= (const int*)d_in[2];
    const float* W0 = (const float*)d_in[3],  *b0  = (const float*)d_in[4];
    const float* W1 = (const float*)d_in[5],  *b1  = (const float*)d_in[6];
    const float* W2 = (const float*)d_in[7],  *b2  = (const float*)d_in[8];
    const float* Wl1= (const float*)d_in[9],  *bl1 = (const float*)d_in[10];
    const float* Wl2= (const float*)d_in[11], *bl2 = (const float*)d_in[12];
    const float* Wl3= (const float*)d_in[13], *bl3 = (const float*)d_in[14];
    float* out = (float*)d_out;

    // workspace layout (floats)
    float* ws = (float*)d_ws;
    float* dinv   = ws;                                  // 50000
    float* h      = ws + 50048;                          // 6.4M
    float* act    = h  + (size_t)N_NODES * 128;          // 6.4M
    float* gmaxf  = act + (size_t)N_NODES * 128;         // 131072
    float* gsum   = gmaxf + N_GRAPHS * 128;              // 131072
    float* counts = gsum  + N_GRAPHS * 128;              // 1024

    const int T = 256;
    // degree -> dinv
    zero_f32<<<(N_NODES + T - 1) / T, T, 0, stream>>>(dinv, N_NODES);
    deg_kernel<<<(N_EDGES + T - 1) / T, T, 0, stream>>>(dst, dinv, N_EDGES);
    dinv_kernel<<<(N_NODES + T - 1) / T, T, 0, stream>>>(dinv, N_NODES);

    const int gemmGrid = (ROW_TILES + 7) / 8;            // 8 waves / block
    const int iaGrid   = (N_NODES * 32 + T - 1) / T;
    const int esGrid   = (N_EDGES * 32 + T - 1) / T;

    // layer 0
    gemm_wmma_bf16<false><<<gemmGrid, 256, 0, stream>>>(x, W0, h);
    init_agg<<<iaGrid, T, 0, stream>>>(h, dinv, b0, act);
    edge_scatter<<<esGrid, T, 0, stream>>>(src, dst, h, dinv, act);
    // layer 1 (ReLU fused into GEMM A-load)
    gemm_wmma_bf16<true><<<gemmGrid, 256, 0, stream>>>(act, W1, h);
    init_agg<<<iaGrid, T, 0, stream>>>(h, dinv, b1, act);
    edge_scatter<<<esGrid, T, 0, stream>>>(src, dst, h, dinv, act);
    // layer 2
    gemm_wmma_bf16<true><<<gemmGrid, 256, 0, stream>>>(act, W2, h);
    init_agg<<<iaGrid, T, 0, stream>>>(h, dinv, b2, act);
    edge_scatter<<<esGrid, T, 0, stream>>>(src, dst, h, dinv, act);

    // pooling (ReLU fused into read; gmax init 0 is correct since values >= 0)
    zero_f32<<<(N_GRAPHS * 128 + T - 1) / T, T, 0, stream>>>(gmaxf, N_GRAPHS * 128);
    zero_f32<<<(N_GRAPHS * 128 + T - 1) / T, T, 0, stream>>>(gsum,  N_GRAPHS * 128);
    zero_f32<<<(N_GRAPHS + T - 1) / T, T, 0, stream>>>(counts, N_GRAPHS);
    counts_kernel<<<(N_NODES + T - 1) / T, T, 0, stream>>>(batch, counts);
    pool_kernel<<<(N_NODES * 128 + T - 1) / T, T, 0, stream>>>(
        act, batch, (unsigned int*)gmaxf, gsum);

    // MLP head
    mlp_kernel<<<N_GRAPHS, 128, 0, stream>>>(gmaxf, gsum, counts,
                                             Wl1, bl1, Wl2, bl2, Wl3, bl3, out);
}
